// SelfAttention_23321672417389
// MI455X (gfx1250) — matrix operand order repaired and checked by reading, not compiled, not run on hardware
//
#include <hip/hip_runtime.h>
#include <hip/hip_bf16.h>

// MI455X (gfx1250) wave32 WMMA self-attention pipeline.
// N=4, L=2048, E=512, H=8, D=64. f16 WMMA + fp32 accumulate; softmax fp32 (exp2 domain).
// Attention uses 4-wave workgroups with double-buffered async LDS staging of K/V.

typedef __attribute__((ext_vector_type(16))) _Float16 v16h;
typedef __attribute__((ext_vector_type(8)))  _Float16 v8h;
typedef __attribute__((ext_vector_type(8)))  float    v8f;

#define NB 4
#define LL 2048
#define EE 512
#define HH 8
#define DD 64
#define KC 32          // keys per chunk in attention
#define QW 4           // waves per attention workgroup

__device__ __forceinline__ v8f wmma_f16(v16h a, v16h b, v8f c) {
  // D = A(16x32 f16) * B(32x16 f16) + C(16x16 f32)
  return __builtin_amdgcn_wmma_f32_16x16x32_f16(false, a, false, b, (short)0, c,
                                                false, false);
}

// A-fragment (16x32, f16) from row-major f16.
// lane<16 -> row=lane, K={0..15}; lane>=16 -> row=lane-16, K={16..31}
__device__ __forceinline__ v16h load_a_f16(const _Float16* base, int ld, int row0,
                                           int col0, int lane) {
  int l16 = lane & 15, g = lane >> 4;
  const _Float16* p = base + (size_t)(row0 + l16) * ld + col0 + 16 * g;
  v8h lo = *(const v8h*)p;
  v8h hi = *(const v8h*)(p + 8);
  v16h a;
#pragma unroll
  for (int i = 0; i < 8; ++i) { a[i] = lo[i]; a[i + 8] = hi[i]; }
  return a;
}

// A-fragment from row-major f32 with convert.
__device__ __forceinline__ v16h load_a_f32(const float* base, int ld, int row0,
                                           int col0, int lane) {
  int l16 = lane & 15, g = lane >> 4;
  const float* p = base + (size_t)(row0 + l16) * ld + col0 + 16 * g;
  v16h a;
#pragma unroll
  for (int i = 0; i < 8; ++i) { a[i] = (_Float16)p[i]; a[i + 8] = (_Float16)p[i + 8]; }
  return a;
}

// B-fragment (32x16 f16): per lane, 16 contiguous K values of one column (global, 32B aligned).
__device__ __forceinline__ v16h load_b_f16(const _Float16* p) {
  return *(const v16h*)p;
}

// B-fragment from LDS with only 16B alignment guaranteed (padded rows).
__device__ __forceinline__ v16h load_b_lds(const _Float16* p) {
  v8h lo = *(const v8h*)p;
  v8h hi = *(const v8h*)(p + 8);
  v16h b;
#pragma unroll
  for (int i = 0; i < 8; ++i) { b[i] = lo[i]; b[i + 8] = hi[i]; }
  return b;
}

// gfx1250 async global->LDS copy (16B), tracked by ASYNCcnt.
__device__ __forceinline__ void async_ld_b128(unsigned lds_off, unsigned long long gaddr) {
  asm volatile("global_load_async_to_lds_b128 %0, %1, off"
               :: "v"(lds_off), "v"(gaddr) : "memory");
}
__device__ __forceinline__ void wait_async0() {
  asm volatile("s_wait_asynccnt 0x0" ::: "memory");
}
__device__ __forceinline__ void wait_ds0() {
  asm volatile("s_wait_dscnt 0x0" ::: "memory");
}

__global__ void cvt_f32_to_f16(const float* __restrict__ s, _Float16* __restrict__ d, int n) {
  int i = blockIdx.x * blockDim.x + threadIdx.x;
  if (i < n) d[i] = (_Float16)s[i];
}

// ---------------- Projection: x[n,l,h,:] @ W^T for q,k,v -------------------
// q_h,k_h: [N,H,L,D] row-major f16.  vT_h: [N,H,D,L] f16 (transposed for AV B-frags).
__global__ __launch_bounds__(256) void proj_kernel(
    const float* __restrict__ vals, const float* __restrict__ keys,
    const float* __restrict__ qry,
    const _Float16* __restrict__ Wq, const _Float16* __restrict__ Wk,
    const _Float16* __restrict__ Wv,
    _Float16* __restrict__ qh, _Float16* __restrict__ kh, _Float16* __restrict__ vT) {
  int lane = threadIdx.x & 31;
  int l16 = lane & 15, g = lane >> 4;
  int wid = blockIdx.x * 8 + (threadIdx.x >> 5);   // 0..4095
  int lt = wid & (LL / 16 - 1);
  int nh = wid >> 7;
  int h = nh & (HH - 1);
  int n = nh >> 3;
  int l0 = lt * 16;

  const float* xq = qry  + (size_t)n * LL * EE + h * DD;
  const float* xk = keys + (size_t)n * LL * EE + h * DD;
  const float* xv = vals + (size_t)n * LL * EE + h * DD;

  v16h aq[2], ak[2], av[2];
#pragma unroll
  for (int c = 0; c < 2; ++c) {
    aq[c] = load_a_f32(xq, EE, l0, 32 * c, lane);
    ak[c] = load_a_f32(xk, EE, l0, 32 * c, lane);
    av[c] = load_a_f32(xv, EE, l0, 32 * c, lane);
  }

  _Float16* qdst = qh + (size_t)(n * HH + h) * LL * DD;
  _Float16* kdst = kh + (size_t)(n * HH + h) * LL * DD;
  _Float16* vdst = vT + (size_t)(n * HH + h) * DD * LL;

#pragma unroll
  for (int et = 0; et < 4; ++et) {
    int ecol = et * 16 + l16;
    v8f cq = {}, ck = {}, cv = {};
#pragma unroll
    for (int c = 0; c < 2; ++c) {
      cq = wmma_f16(aq[c], load_b_f16(Wq + ecol * DD + 32 * c + 16 * g), cq);
      ck = wmma_f16(ak[c], load_b_f16(Wk + ecol * DD + 32 * c + 16 * g), ck);
      cv = wmma_f16(av[c], load_b_f16(Wv + ecol * DD + 32 * c + 16 * g), cv);
    }
#pragma unroll
    for (int r = 0; r < 8; ++r) {
      int row = l0 + r + 8 * g;
      qdst[(size_t)row * DD + et * 16 + l16] = (_Float16)cq[r];
      kdst[(size_t)row * DD + et * 16 + l16] = (_Float16)ck[r];
    }
    v8h pk;
#pragma unroll
    for (int r = 0; r < 8; ++r) pk[r] = (_Float16)cv[r];
    *(v8h*)(vdst + (size_t)(et * 16 + l16) * LL + l0 + 8 * g) = pk;
  }
}

// ---------------- Flash attention ----------------------------------------
// Block = 128 threads (4 waves) = 64 queries of one (n,h).
// K/V tiles staged to LDS with async b128 copies, double-buffered.
__global__ __launch_bounds__(128) void attn_kernel(
    const _Float16* __restrict__ qh, const _Float16* __restrict__ kh,
    const _Float16* __restrict__ vT, const int* __restrict__ mask,
    _Float16* __restrict__ X) {
  // Padded rows: K stride 72 halves (bank stride 36), V/P stride 40 (bank stride 20).
  __shared__ __align__(16) _Float16 kt[2][KC][72];     // [buf][key][d]
  __shared__ __align__(16) _Float16 vt[2][DD][40];     // [buf][d][key]
  __shared__ __align__(16) _Float16 pt[QW][16][40];    // per-wave P relayout

  int tid = threadIdx.x;
  int lane = tid & 31;
  int w = tid >> 5;                 // wave in block
  int l16 = lane & 15, g = lane >> 4;

  int bid = blockIdx.x;             // N*H*(L/64) = 1024
  int qt = bid & (LL / 64 - 1);
  int nh = bid >> 5;
  int h = nh & (HH - 1), n = nh >> 3;
  int q0 = qt * 64 + w * 16;

  const _Float16* qb = qh + (size_t)(n * HH + h) * LL * DD;
  const _Float16* kb = kh + (size_t)(n * HH + h) * LL * DD;
  const _Float16* vb = vT + (size_t)(n * HH + h) * DD * LL;
  const int* mb = mask + (size_t)n * LL * LL;

  // Per-thread staging tasks: K = 32 rows x 4 segs of 32B; V = 64 rows x 2 segs of 32B.
  int krow = tid >> 2, kseg = tid & 3;
  int vrow = tid >> 1, vseg = tid & 1;
  unsigned long long kga = (unsigned long long)(kb + (size_t)krow * DD + kseg * 16);
  unsigned long long vga = (unsigned long long)(vb + (size_t)vrow * LL + vseg * 16);
  unsigned kl[2], vl[2];
  kl[0] = (unsigned)(uintptr_t)&kt[0][krow][kseg * 16];
  kl[1] = (unsigned)(uintptr_t)&kt[1][krow][kseg * 16];
  vl[0] = (unsigned)(uintptr_t)&vt[0][vrow][vseg * 16];
  vl[1] = (unsigned)(uintptr_t)&vt[1][vrow][vseg * 16];

  v16h aq0 = load_a_f16(qb, DD, q0, 0, lane);
  v16h aq1 = load_a_f16(qb, DD, q0, 32, lane);

  // exp2-domain softmax: scale = log2(e)/sqrt(512)
  const float SCALE2 = 0.06376258953f;
  const float NEG2 = -6.376258953e18f;   // -1e20 * SCALE2

  float mi[8], li[8];
  v8f o0 = {}, o1 = {}, o2 = {}, o3 = {};
#pragma unroll
  for (int r = 0; r < 8; ++r) { mi[r] = -3.0e38f; li[r] = 0.0f; }

  // Prologue: stage chunk 0 into buffer 0.
  {
    async_ld_b128(kl[0], kga);      async_ld_b128(kl[0] + 16, kga + 16);
    async_ld_b128(vl[0], vga);      async_ld_b128(vl[0] + 16, vga + 16);
  }
  wait_async0();
  __syncthreads();

  const int NIT = LL / KC;
  for (int it = 0; it < NIT; ++it) {
    int kc = it * KC;
    int buf = it & 1;
    // Prefetch next chunk into the other buffer (overlaps with compute below).
    if (it + 1 < NIT) {
      unsigned long long kg = kga + (unsigned long long)(kc + KC) * DD * 2;
      unsigned long long vg = vga + (unsigned long long)(kc + KC) * 2;
      async_ld_b128(kl[buf ^ 1], kg);      async_ld_b128(kl[buf ^ 1] + 16, kg + 16);
      async_ld_b128(vl[buf ^ 1], vg);      async_ld_b128(vl[buf ^ 1] + 16, vg + 16);
    }

    // ---- S = Q K^T for 32 keys (two 16-col tiles) from LDS ----
    v8f s[2];
#pragma unroll
    for (int t = 0; t < 2; ++t) {
      const _Float16* kp = &kt[buf][16 * t + l16][16 * g];
      v8f acc = {};
      acc = wmma_f16(aq0, load_b_lds(kp), acc);        // d = 0..31
      acc = wmma_f16(aq1, load_b_lds(kp + 32), acc);   // d = 32..63
      s[t] = acc;
    }
    // ---- mask + scale (lane = key column -> coalesced mask rows) ----
#pragma unroll
    for (int t = 0; t < 2; ++t) {
      int kcol = kc + 16 * t + l16;
#pragma unroll
      for (int r = 0; r < 8; ++r) {
        int qrow = q0 + r + 8 * g;
        int mv = mb[(size_t)qrow * LL + kcol];
        float e = s[t][r] * SCALE2;
        s[t][r] = (mv == 0) ? NEG2 : e;
      }
    }
    // ---- online softmax (rows live across the 16 lanes of each half) ----
    float alpha[8];
#pragma unroll
    for (int r = 0; r < 8; ++r) {
      float v = fmaxf(s[0][r], s[1][r]);
#pragma unroll
      for (int off = 8; off >= 1; off >>= 1) v = fmaxf(v, __shfl_xor(v, off, 32));
      float mn = fmaxf(mi[r], v);
      alpha[r] = __builtin_exp2f(mi[r] - mn);
      mi[r] = mn;
    }
#pragma unroll
    for (int t = 0; t < 2; ++t)
#pragma unroll
      for (int r = 0; r < 8; ++r) s[t][r] = __builtin_exp2f(s[t][r] - mi[r]);
#pragma unroll
    for (int r = 0; r < 8; ++r) {
      float v = s[0][r] + s[1][r];
#pragma unroll
      for (int off = 8; off >= 1; off >>= 1) v += __shfl_xor(v, off, 32);
      li[r] = li[r] * alpha[r] + v;
    }
#pragma unroll
    for (int r = 0; r < 8; ++r) {
      o0[r] *= alpha[r]; o1[r] *= alpha[r]; o2[r] *= alpha[r]; o3[r] *= alpha[r];
    }
    // ---- P: C-frag layout -> A-frag layout via per-wave LDS slice ----
#pragma unroll
    for (int t = 0; t < 2; ++t)
#pragma unroll
      for (int r = 0; r < 8; ++r)
        pt[w][r + 8 * g][16 * t + l16] = (_Float16)s[t][r];
    wait_ds0();                         // per-wave DS ops are in-order; wait for stores
    v16h ap;
    {
      v8h lo = *(const v8h*)&pt[w][l16][16 * g];
      v8h hi = *(const v8h*)&pt[w][l16][16 * g + 8];
#pragma unroll
      for (int i = 0; i < 8; ++i) { ap[i] = lo[i]; ap[i + 8] = hi[i]; }
    }
    // ---- O += P V from LDS (V transposed: column contiguous over keys) ----
    const _Float16* vp = &vt[buf][l16][16 * g];
    o0 = wmma_f16(ap, load_b_lds(vp), o0);
    o1 = wmma_f16(ap, load_b_lds(vp + 16 * 40), o1);
    o2 = wmma_f16(ap, load_b_lds(vp + 32 * 40), o2);
    o3 = wmma_f16(ap, load_b_lds(vp + 48 * 40), o3);

    // Next buffer ready + everyone done with current buffer.
    wait_async0();
    __syncthreads();
  }
  // ---- normalize + store to X [N,L,E] f16 ----
#pragma unroll
  for (int r = 0; r < 8; ++r) li[r] = 1.0f / li[r];
  _Float16* xb = X + (size_t)n * LL * EE + h * DD;
#pragma unroll
  for (int r = 0; r < 8; ++r) {
    int row = q0 + r + 8 * g;
    _Float16* xp = xb + (size_t)row * EE;
    xp[0 * 16 + l16] = (_Float16)(o0[r] * li[r]);
    xp[1 * 16 + l16] = (_Float16)(o1[r] * li[r]);
    xp[2 * 16 + l16] = (_Float16)(o2[r] * li[r]);
    xp[3 * 16 + l16] = (_Float16)(o3[r] * li[r]);
  }
}

// ---------------- Output projection: out = X @ Wo^T + bo ------------------
__global__ __launch_bounds__(256) void outproj_kernel(
    const _Float16* __restrict__ X, const _Float16* __restrict__ Wo,
    const float* __restrict__ bo, float* __restrict__ out) {
  int lane = threadIdx.x & 31;
  int l16 = lane & 15, g = lane >> 4;
  int wid = blockIdx.x * 8 + (threadIdx.x >> 5);  // 0..4095
  int mt = wid >> 3;                              // 0..511
  int nt = wid & 7;                               // 0..7
  int m0 = mt * 16;
  int e0 = nt * 64;
  v8f a0 = {}, a1 = {}, a2 = {}, a3 = {};
  for (int kc = 0; kc < EE; kc += 32) {
    v16h a = load_a_f16(X, EE, m0, kc, lane);
    a0 = wmma_f16(a, load_b_f16(Wo + (size_t)(e0 + 0 * 16 + l16) * EE + kc + 16 * g), a0);
    a1 = wmma_f16(a, load_b_f16(Wo + (size_t)(e0 + 1 * 16 + l16) * EE + kc + 16 * g), a1);
    a2 = wmma_f16(a, load_b_f16(Wo + (size_t)(e0 + 2 * 16 + l16) * EE + kc + 16 * g), a2);
    a3 = wmma_f16(a, load_b_f16(Wo + (size_t)(e0 + 3 * 16 + l16) * EE + kc + 16 * g), a3);
  }
#pragma unroll
  for (int t = 0; t < 4; ++t) {
    float bv = bo[e0 + t * 16 + l16];
    v8f c = (t == 0) ? a0 : (t == 1) ? a1 : (t == 2) ? a2 : a3;
#pragma unroll
    for (int r = 0; r < 8; ++r) {
      int row = m0 + r + 8 * g;
      out[(size_t)row * EE + e0 + t * 16 + l16] = c[r] + bv;
    }
  }
}

// ---------------- Host launcher -------------------------------------------
extern "C" void kernel_launch(void* const* d_in, const int* in_sizes, int n_in,
                              void* d_out, int out_size, void* d_ws, size_t ws_size,
                              hipStream_t stream) {
  // setup_inputs order: values, keys, query, mask, Wv, Wk, Wq, Wo, bo
  const float* vals = (const float*)d_in[0];
  const float* keys = (const float*)d_in[1];
  const float* qry  = (const float*)d_in[2];
  const int*   mask = (const int*)d_in[3];
  const float* Wv = (const float*)d_in[4];
  const float* Wk = (const float*)d_in[5];
  const float* Wq = (const float*)d_in[6];
  const float* Wo = (const float*)d_in[7];
  const float* bo = (const float*)d_in[8];
  float* out = (float*)d_out;

  // Workspace layout (f16 elements). Total ~32.6 MB.
  _Float16* ws = (_Float16*)d_ws;
  _Float16* Wq_h = ws + 0;          //  64*64
  _Float16* Wk_h = ws + 4096;       //  64*64
  _Float16* Wv_h = ws + 8192;       //  64*64
  _Float16* Wo_h = ws + 16384;      // 512*512
  _Float16* q_h  = ws + 278528;     // N*H*L*D
  _Float16* k_h  = ws + 4472832;    // N*H*L*D
  _Float16* vT_h = ws + 8667136;    // N*H*D*L
  _Float16* X_h  = ws + 12861440;   // N*L*E

  cvt_f32_to_f16<<<16, 256, 0, stream>>>(Wq, Wq_h, 64 * 64);
  cvt_f32_to_f16<<<16, 256, 0, stream>>>(Wk, Wk_h, 64 * 64);
  cvt_f32_to_f16<<<16, 256, 0, stream>>>(Wv, Wv_h, 64 * 64);
  cvt_f32_to_f16<<<1024, 256, 0, stream>>>(Wo, Wo_h, 512 * 512);

  proj_kernel<<<512, 256, 0, stream>>>(vals, keys, qry, Wq_h, Wk_h, Wv_h,
                                       q_h, k_h, vT_h);
  attn_kernel<<<NB * HH * (LL / 64), 128, 0, stream>>>(q_h, k_h, vT_h, mask, X_h);
  outproj_kernel<<<512, 256, 0, stream>>>(X_h, Wo_h, bo, out);
}
